// InstanceHead_28836410426048
// MI455X (gfx1250) — matrix operand order repaired and checked by reading, not compiled, not run on hardware
//
#include <hip/hip_runtime.h>

// ---------------------------------------------------------------------------
// MI455X (gfx1250) implementation.
//   conv0/conv1 (the 1.2e11-FLOP hot path) -> implicit GEMM via
//   v_wmma_f32_16x16x32_bf16 (bf16 operands, f32 accumulate).
//   Each wave owns a 32Mx32N register tile (4 accumulators) so the LDS
//   ds_load_b128 : wmma ratio is 2:1 (LDS bandwidth balanced vs matrix pipe).
//   Per-tap weight slices are software-pipelined through *named* staging
//   registers (no local array -> no promote-to-LDS alloca) into ping-pong
//   LDS buffers; outputs bounce through LDS for coalesced b128 stores.
// ---------------------------------------------------------------------------

typedef __attribute__((ext_vector_type(16))) __bf16 v16bf;
typedef __attribute__((ext_vector_type(8)))  __bf16 v8bf;
typedef __attribute__((ext_vector_type(8)))  float  v8f;

__device__ __forceinline__ __bf16 f2bf(float f) {
  union { float f; unsigned u; } in; in.f = f;
  unsigned u = in.u;
  unsigned r = (u + 0x7fffu + ((u >> 16) & 1u)) >> 16;   // RNE truncate
  union { unsigned short s; __bf16 h; } out; out.s = (unsigned short)r;
  return out.h;
}
__device__ __forceinline__ float sigmoidf(float x) { return 1.0f / (1.0f + __expf(-x)); }

__device__ __forceinline__ v16bf cat8(v8bf lo, v8bf hi) {
  return __builtin_shufflevector(lo, hi, 0,1,2,3,4,5,6,7,8,9,10,11,12,13,14,15);
}

// Stage 8 input channels -> 8 bf16 in LDS (16B store).
__device__ __forceinline__ void load8(const float* __restrict__ src, __bf16* dst) {
  float4 a = *(const float4*)(src);
  float4 c = *(const float4*)(src + 4);
  v8bf t;
  t[0]=f2bf(a.x); t[1]=f2bf(a.y); t[2]=f2bf(a.z); t[3]=f2bf(a.w);
  t[4]=f2bf(c.x); t[5]=f2bf(c.y); t[6]=f2bf(c.z); t[7]=f2bf(c.w);
  *(v8bf*)dst = t;
}
__device__ __forceinline__ void load8(const __bf16* __restrict__ src, __bf16* dst) {
  *(int4*)dst = *(const int4*)src;
}

// ---------------------------------------------------------------------------
// Weight prep: transpose HWIO conv weights [K=9*Cin][64] -> Bt[64][K] bf16 so
// each WMMA B-operand lane reads K-contiguous 16B chunks for its column.
// ---------------------------------------------------------------------------
__global__ __launch_bounds__(256)
void prep_weights(const float* __restrict__ w0, const float* __restrict__ w1,
                  __bf16* __restrict__ Bt0, __bf16* __restrict__ Bt1) {
  int i = blockIdx.x * 256 + threadIdx.x;
  const int T0 = 64 * 288, T1 = 64 * 576;
  if (i < T0) {
    int n = i / 288, k = i % 288;
    Bt0[i] = f2bf(w0[k * 64 + n]);
  } else if (i < T0 + T1) {
    int j = i - T0;
    int n = j / 576, k = j % 576;
    Bt1[j] = f2bf(w1[k * 64 + n]);
  }
}

// ---------------------------------------------------------------------------
// Fused 3x3 conv + bias + ReLU + inference-BN, implicit GEMM with WMMA.
// Block: 128 thr = 4 waves in a 2Mx2N grid; block tile = 64 pixels x 64 chan.
// Wave (wm,wn): rows wm*32..+32 (two 16-row groups), cols wn*32..+32 (two
// 16-col groups) -> 4 accumulators, 8 ds_loads : 4 wmma per K-step.
// K = 9 taps * CIN.  Input halo [3][66][CIN] bf16 staged once; per-tap B
// slice [64][CIN] pipelined: tap t+1 global loads issue before tap t WMMAs,
// commit to the alternate LDS buffer after them (one barrier per tap).
// Output bounces through LDS for fully coalesced b128 stores.
// ---------------------------------------------------------------------------
template <int CIN, typename TIN>
__global__ __launch_bounds__(128)
void conv3x3_bnrelu_wmma(const TIN* __restrict__ in,      // [4,512,512,CIN]
                         const __bf16* __restrict__ Bt,   // [64][9*CIN]
                         const float* __restrict__ bias,
                         const float* __restrict__ gam, const float* __restrict__ bet,
                         const float* __restrict__ mu,  const float* __restrict__ var,
                         __bf16* __restrict__ out)        // [4,512,512,64] bf16
{
  constexpr int H = 512, W = 512, COUT = 64, K = 9 * CIN;
  constexpr int XT = 64, HWD = XT + 2;      // 66 halo x-positions
  constexpr int CINP = CIN + 8;             // padded row -> conflict-free LDS
  constexpr int C8 = CIN / 8;
  constexpr int NT = 128;                   // threads per block
  constexpr int NSTG = (COUT * C8) / NT;    // int4 staging regs per thread
  static_assert(NSTG == 2 || NSTG == 4, "staging layout");

  __shared__ __align__(16) __bf16 halo[3 * HWD * CINP];
  __shared__ __align__(16) __bf16 Bs[2][COUT * CINP];
  __shared__ __align__(16) __bf16 outbuf[XT * 72];   // 64 px x 64ch (+pad)

  const int tid = threadIdx.x;
  const int bx = blockIdx.x % (W / XT);
  const int y  = (blockIdx.x / (W / XT)) % H;
  const int b  = blockIdx.x / ((W / XT) * H);
  const int x0 = bx * XT;

  // ---- stage halo, fp32/bf16 -> bf16 ----
  for (int t = tid; t < 3 * HWD * C8; t += NT) {
    int c  = (t % C8) * 8;
    int pr = t / C8;            // 0..197
    int rx = pr % HWD;          // 0..65
    int ry = pr / HWD;          // 0..2
    int iy = y + ry - 1;
    int ix = x0 + rx - 1;
    __bf16* dst = &halo[(ry * HWD + rx) * CINP + c];
    if (iy >= 0 && iy < H && ix >= 0 && ix < W) {
      load8(&in[((size_t)(b * H + iy) * W + ix) * CIN + c], dst);
    } else {
      int4 z = {0, 0, 0, 0};
      *(int4*)dst = z;
    }
  }

  const int wave = tid >> 5, lane = tid & 31;
  const int wm = wave & 1, wn = wave >> 1;
  const int lhalf = lane >> 4, l15 = lane & 15;

  // ---- per-thread staging offsets (tap-0 global offset; LDS offset) ----
  const int t0 = tid;
  const int t1 = tid + NT;
  const size_t bt0 = (size_t)(t0 / C8) * K + (t0 % C8) * 8;
  const size_t bt1 = (size_t)(t1 / C8) * K + (t1 % C8) * 8;
  const int    bs0 = (t0 / C8) * CINP + (t0 % C8) * 8;
  const int    bs1 = (t1 / C8) * CINP + (t1 % C8) * 8;
  size_t bt2 = 0, bt3 = 0; int bs2 = 0, bs3 = 0;
  if constexpr (NSTG == 4) {
    const int t2 = tid + 2 * NT;
    const int t3 = tid + 3 * NT;
    bt2 = (size_t)(t2 / C8) * K + (t2 % C8) * 8;
    bt3 = (size_t)(t3 / C8) * K + (t3 % C8) * 8;
    bs2 = (t2 / C8) * CINP + (t2 % C8) * 8;
    bs3 = (t3 / C8) * CINP + (t3 % C8) * 8;
  }

  // named staging registers (no array -> stays in VGPRs, no LDS alloca)
  int4 stg0 = {0,0,0,0}, stg1 = {0,0,0,0}, stg2 = {0,0,0,0}, stg3 = {0,0,0,0};

  // prologue: fetch + commit tap 0 into Bs[0]
  stg0 = *(const int4*)&Bt[bt0];
  stg1 = *(const int4*)&Bt[bt1];
  if constexpr (NSTG == 4) {
    stg2 = *(const int4*)&Bt[bt2];
    stg3 = *(const int4*)&Bt[bt3];
  }
  *(int4*)&Bs[0][bs0] = stg0;
  *(int4*)&Bs[0][bs1] = stg1;
  if constexpr (NSTG == 4) {
    *(int4*)&Bs[0][bs2] = stg2;
    *(int4*)&Bs[0][bs3] = stg3;
  }
  __syncthreads();                           // halo + Bs[0] ready

  v8f acc00 = {}, acc01 = {}, acc10 = {}, acc11 = {};

  for (int tap = 0; tap < 9; ++tap) {
    if (tap < 8) {                           // issue next tap's loads, no wait
      const size_t to = (size_t)(tap + 1) * CIN;
      stg0 = *(const int4*)&Bt[bt0 + to];
      stg1 = *(const int4*)&Bt[bt1 + to];
      if constexpr (NSTG == 4) {
        stg2 = *(const int4*)&Bt[bt2 + to];
        stg3 = *(const int4*)&Bt[bt3 + to];
      }
    }

    const int dy = tap / 3, dx = tap % 3;
    const __bf16* ab0 = &halo[(dy * HWD + (wm * 32 + l15 + dx)) * CINP];
    const __bf16* ab1 = ab0 + 16 * CINP;     // rows +16
    const __bf16* bsl = Bs[tap & 1];

#pragma unroll
    for (int ks = 0; ks < CIN / 32; ++ks) {
      const int cb = ks * 32;
      // A 16x32 bf16: lanes 0-15 hold K {0..7,16..23}, lanes 16-31 {8..15,24..31}
      v16bf a0 = cat8(*(const v8bf*)(ab0 + cb + lhalf * 8),
                      *(const v8bf*)(ab0 + cb + lhalf * 8 + 16));
      v16bf a1 = cat8(*(const v8bf*)(ab1 + cb + lhalf * 8),
                      *(const v8bf*)(ab1 + cb + lhalf * 8 + 16));
      // B 32x16 bf16: lane = column (n), lanes 0-15 hold K {0..7,16..23}, lanes 16-31 {8..15,24..31}
      const __bf16* bb0 = &bsl[(wn * 32 + l15) * CINP + cb + lhalf * 8];
      const __bf16* bb1 = &bsl[(wn * 32 + 16 + l15) * CINP + cb + lhalf * 8];
      v16bf bm0 = cat8(*(const v8bf*)(bb0), *(const v8bf*)(bb0 + 16));
      v16bf bm1 = cat8(*(const v8bf*)(bb1), *(const v8bf*)(bb1 + 16));
      acc00 = __builtin_amdgcn_wmma_f32_16x16x32_bf16(false, a0, false, bm0,
                                                      (short)0, acc00, false, false);
      acc01 = __builtin_amdgcn_wmma_f32_16x16x32_bf16(false, a0, false, bm1,
                                                      (short)0, acc01, false, false);
      acc10 = __builtin_amdgcn_wmma_f32_16x16x32_bf16(false, a1, false, bm0,
                                                      (short)0, acc10, false, false);
      acc11 = __builtin_amdgcn_wmma_f32_16x16x32_bf16(false, a1, false, bm1,
                                                      (short)0, acc11, false, false);
    }

    if (tap < 8) {                           // loadcnt wait lands behind WMMAs
      __bf16* bp = Bs[(tap + 1) & 1];
      *(int4*)&bp[bs0] = stg0;
      *(int4*)&bp[bs1] = stg1;
      if constexpr (NSTG == 4) {
        *(int4*)&bp[bs2] = stg2;
        *(int4*)&bp[bs3] = stg3;
      }
    }
    __syncthreads();
  }

  // ---- epilogue: bias + relu + BN, then LDS bounce for coalesced stores ----
  // C/D layout: VGPR v -> M = v + 8*(lane>=16), N = lane&15.
  {
    const int n0 = wn * 32 + l15;
    const int n1 = n0 + 16;
    const float bi0 = bias[n0], bi1 = bias[n1];
    const float sc0 = gam[n0] * rsqrtf(var[n0] + 1e-3f);
    const float sh0 = bet[n0] - mu[n0] * sc0;
    const float sc1 = gam[n1] * rsqrtf(var[n1] + 1e-3f);
    const float sh1 = bet[n1] - mu[n1] * sc1;
#pragma unroll
    for (int v = 0; v < 8; ++v) {
      const int r0 = wm * 32 + v + lhalf * 8;          // rows 0..15 group
      const int r1 = r0 + 16;                          // rows 16..31 group
      outbuf[r0 * 72 + n0] = f2bf(fmaxf(acc00[v] + bi0, 0.0f) * sc0 + sh0);
      outbuf[r0 * 72 + n1] = f2bf(fmaxf(acc01[v] + bi1, 0.0f) * sc1 + sh1);
      outbuf[r1 * 72 + n0] = f2bf(fmaxf(acc10[v] + bi0, 0.0f) * sc0 + sh0);
      outbuf[r1 * 72 + n1] = f2bf(fmaxf(acc11[v] + bi1, 0.0f) * sc1 + sh1);
    }
  }
  __syncthreads();
  // 64 px x 64 ch x 2B = 8KB, 512 b128 chunks -> fully coalesced stores
  for (int t = tid; t < XT * COUT / 8; t += NT) {
    const int p = t >> 3;
    const int c = (t & 7) * 8;
    const size_t base = ((size_t)(b * H + y) * W + (x0 + p)) * COUT + c;
    *(int4*)&out[base] = *(const int4*)&outbuf[p * 72 + c];
  }
}

// ---------------------------------------------------------------------------
// 1x1 conv 64->4 + bias + ReLU + BN (negligible FLOPs, bandwidth bound).
// ---------------------------------------------------------------------------
__global__ __launch_bounds__(256)
void convl_kernel(const __bf16* __restrict__ x1, const float* __restrict__ w,
                  const float* __restrict__ bias,
                  const float* __restrict__ gam, const float* __restrict__ bet,
                  const float* __restrict__ mu,  const float* __restrict__ var,
                  float* __restrict__ x2)
{
  __shared__ float ws_[64 * 4];
  __shared__ float psc[4], psh[4], pb[4];
  const int tid = threadIdx.x;
  ws_[tid] = w[tid];
  if (tid < 4) {
    float s = gam[tid] * rsqrtf(var[tid] + 1e-3f);
    psc[tid] = s; psh[tid] = bet[tid] - mu[tid] * s; pb[tid] = bias[tid];
  }
  __syncthreads();
  const size_t pix = (size_t)blockIdx.x * 256 + tid;   // < 4*512*512 exactly
  const __bf16* p = x1 + pix * 64;
  float acc[4] = {0.f, 0.f, 0.f, 0.f};
#pragma unroll
  for (int c8 = 0; c8 < 8; ++c8) {
    int4 raw = *(const int4*)(p + c8 * 8);
    const unsigned short* hs = (const unsigned short*)&raw;
#pragma unroll
    for (int j = 0; j < 8; ++j) {
      union { unsigned u; float f; } cv; cv.u = ((unsigned)hs[j]) << 16;
      const float xv = cv.f;
      const int c = c8 * 8 + j;
#pragma unroll
      for (int n = 0; n < 4; ++n) acc[n] += xv * ws_[c * 4 + n];
    }
  }
  float4 o;
  o.x = fmaxf(acc[0] + pb[0], 0.f) * psc[0] + psh[0];
  o.y = fmaxf(acc[1] + pb[1], 0.f) * psc[1] + psh[1];
  o.z = fmaxf(acc[2] + pb[2], 0.f) * psc[2] + psh[2];
  o.w = fmaxf(acc[3] + pb[3], 0.f) * psc[3] + psh[3];
  *(float4*)(x2 + pix * 4) = o;
}

// ---------------------------------------------------------------------------
// att_a = center @ atta_w + atta_b  (512x256x4 GEMV; one block per instance,
// one wave per output channel, wave32 shfl reduction).
// ---------------------------------------------------------------------------
__global__ __launch_bounds__(128)
void att_center_kernel(const float* __restrict__ lr, const float* __restrict__ loc,
                       const float* __restrict__ aw, const float* __restrict__ ab,
                       float* __restrict__ atta)
{
  const int m = blockIdx.x;           // 0..511
  const int b = m / 128;
  const float ly = fminf(fmaxf(loc[m * 2 + 0], 0.f), 1.f);
  const float lx = fminf(fmaxf(loc[m * 2 + 1], 0.f), 1.f);
  const int iy = min((int)(ly * 128.f + 0.5f), 127);
  const int ix = min((int)(lx * 128.f + 0.5f), 127);
  const float* c = lr + ((size_t)(b * 128 + iy) * 128 + ix) * 256;
  const int wave = threadIdx.x >> 5, lane = threadIdx.x & 31;
  float s = 0.f;
  for (int k = lane; k < 256; k += 32) s += c[k] * aw[k * 4 + wave];
#pragma unroll
  for (int off = 16; off > 0; off >>= 1) s += __shfl_down(s, off, 32);
  if (lane == 0) atta[m * 4 + wave] = s + ab[wave];
}

// ---------------------------------------------------------------------------
// Fused: patch gather from x2, positional attention, sigmoid gate, 1x1x4->1
// output conv, sigmoid, + coords. One thread per output pixel.
// ---------------------------------------------------------------------------
__global__ __launch_bounds__(256)
void final_kernel(const float* __restrict__ x2, const float* __restrict__ loc,
                  const float* __restrict__ atta,
                  const float* __restrict__ abw, const float* __restrict__ abb,
                  const float* __restrict__ ow,  const float* __restrict__ ob,
                  float* __restrict__ out, float* __restrict__ coords)
{
  const int PS = 96;
  const size_t pid = (size_t)blockIdx.x * 256 + threadIdx.x;
  if (pid >= (size_t)512 * PS * PS) return;
  const int m  = (int)(pid / (PS * PS));
  const int p  = (int)(pid % (PS * PS));
  const int py = p / PS, px = p % PS;
  const int b  = m / 128;

  const float ly = fminf(fmaxf(loc[m * 2 + 0], 0.f), 1.f);
  const float lx = fminf(fmaxf(loc[m * 2 + 1], 0.f), 1.f);
  const int cy = (int)(ly * 512.f);
  const int cx = (int)(lx * 512.f);
  const int ii = cy + (py - PS / 2);
  const int jj = cx + (px - PS / 2);
  const int iic = min(max(ii, 0), 511);
  const int jjc = min(max(jj, 0), 511);

  float4 patch = *(const float4*)(x2 + ((size_t)(b * 512 + iic) * 512 + jjc) * 4);
  float4 aa = *(const float4*)(atta + m * 4);

  const float step = 2.0f / 95.0f;
  const float yv = -1.0f + py * step;
  const float xv = -1.0f + px * step;
  const float pe[6] = {yv, xv, yv * yv, xv * xv, yv * yv * yv, xv * xv * xv};

  float att[4];
#pragma unroll
  for (int n = 0; n < 4; ++n) {
    float s = abb[n];
#pragma unroll
    for (int k = 0; k < 6; ++k) s += pe[k] * abw[k * 4 + n];
    att[n] = s;
  }
  const float g0 = sigmoidf(aa.x + att[0]);
  const float g1 = sigmoidf(aa.y + att[1]);
  const float g2 = sigmoidf(aa.z + att[2]);
  const float g3 = sigmoidf(aa.w + att[3]);

  float o = patch.x * g0 * ow[0] + patch.y * g1 * ow[1] +
            patch.z * g2 * ow[2] + patch.w * g3 * ow[3] + ob[0];
  out[pid] = sigmoidf(o);
  coords[pid * 2 + 0] = (float)ii;
  coords[pid * 2 + 1] = (float)jj;
}

// ---------------------------------------------------------------------------
extern "C" void kernel_launch(void* const* d_in, const int* in_sizes, int n_in,
                              void* d_out, int out_size, void* d_ws, size_t ws_size,
                              hipStream_t stream) {
  const float* hr   = (const float*)d_in[0];
  const float* lr   = (const float*)d_in[1];
  const float* loc  = (const float*)d_in[2];
  const float* c0w  = (const float*)d_in[3];
  const float* c0b  = (const float*)d_in[4];
  const float* g0   = (const float*)d_in[5];
  const float* bb0  = (const float*)d_in[6];
  const float* m0   = (const float*)d_in[7];
  const float* v0   = (const float*)d_in[8];
  const float* c1w  = (const float*)d_in[9];
  const float* c1b  = (const float*)d_in[10];
  const float* g1   = (const float*)d_in[11];
  const float* bb1  = (const float*)d_in[12];
  const float* m1   = (const float*)d_in[13];
  const float* v1   = (const float*)d_in[14];
  const float* clw  = (const float*)d_in[15];
  const float* clb  = (const float*)d_in[16];
  const float* gl   = (const float*)d_in[17];
  const float* bbl  = (const float*)d_in[18];
  const float* ml   = (const float*)d_in[19];
  const float* vl   = (const float*)d_in[20];
  const float* ataw = (const float*)d_in[21];
  const float* atab = (const float*)d_in[22];
  const float* atbw = (const float*)d_in[23];
  const float* atbb = (const float*)d_in[24];
  const float* outw = (const float*)d_in[25];
  const float* outb = (const float*)d_in[26];

  // workspace carve-up (~285 MB)
  char* ws = (char*)d_ws;
  size_t off = 0;
  auto carve = [&](size_t bytes) -> void* {
    void* p = ws + off;
    off = (off + bytes + 255) & ~(size_t)255;
    return p;
  };
  __bf16* x0   = (__bf16*)carve((size_t)4 * 512 * 512 * 64 * 2);
  __bf16* x1   = (__bf16*)carve((size_t)4 * 512 * 512 * 64 * 2);
  float*  x2   = (float*) carve((size_t)4 * 512 * 512 * 4 * 4);
  __bf16* Bt0  = (__bf16*)carve((size_t)64 * 288 * 2);
  __bf16* Bt1  = (__bf16*)carve((size_t)64 * 576 * 2);
  float*  atta = (float*) carve((size_t)512 * 4 * 4);
  (void)ws_size; (void)in_sizes; (void)n_in;

  float* out_p    = (float*)d_out;                    // [512*96*96]
  float* coords_p = out_p + (size_t)512 * 96 * 96;    // [512*96*96*2]
  (void)out_size;

  prep_weights<<<216, 256, 0, stream>>>(c0w, c1w, Bt0, Bt1);

  const int conv_blocks = 4 * 512 * (512 / 64);       // 16384
  conv3x3_bnrelu_wmma<32, float><<<conv_blocks, 128, 0, stream>>>(
      hr, Bt0, c0b, g0, bb0, m0, v0, x0);
  conv3x3_bnrelu_wmma<64, __bf16><<<conv_blocks, 128, 0, stream>>>(
      x0, Bt1, c1b, g1, bb1, m1, v1, x1);

  convl_kernel<<<4 * 512 * 512 / 256, 256, 0, stream>>>(
      x1, clw, clb, gl, bbl, ml, vl, x2);

  att_center_kernel<<<512, 128, 0, stream>>>(lr, loc, ataw, atab, atta);

  final_kernel<<<(512 * 96 * 96) / 256, 256, 0, stream>>>(
      x2, loc, atta, atbw, atbb, outw, outb, out_p, coords_p);
}
